// LGCore_39556648796682
// MI455X (gfx1250) — matrix operand order repaired and checked by reading, not compiled, not run on hardware
//
#include <hip/hip_runtime.h>

typedef __attribute__((ext_vector_type(16))) __bf16 v16bf;
typedef __attribute__((ext_vector_type(8)))  float  v8f;

union Frag16 { unsigned u[8]; uint4 q[2]; v16bf v; };
union FragC  { float    f[8]; v8f v; };

__device__ inline unsigned short f2bf_rne(float f) {
    unsigned u = __float_as_uint(f);
    unsigned lsb = (u >> 16) & 1u;
    u += 0x7fffu + lsb;
    return (unsigned short)(u >> 16);
}
__device__ inline float bf2f(unsigned short h) {
    return __uint_as_float(((unsigned)h) << 16);
}
// Split two fp32 into packed (hi,hi) and (lo,lo) bf16 dwords (lo = residual).
__device__ inline void split2(float x0, float x1, unsigned& dhi, unsigned& dlo) {
    unsigned short h0 = f2bf_rne(x0), h1 = f2bf_rne(x1);
    float r0 = x0 - bf2f(h0), r1 = x1 - bf2f(h1);
    unsigned short l0 = f2bf_rne(r0), l1 = f2bf_rne(r1);
    dhi = (unsigned)h0 | ((unsigned)h1 << 16);
    dlo = (unsigned)l0 | ((unsigned)l1 << 16);
}

// ---------------- small utility kernels ----------------
__global__ void fill_kernel(float* p, float v, int n) {
    int i = blockIdx.x * blockDim.x + threadIdx.x;
    if (i < n) p[i] = v;
}

__global__ void deg_count_kernel(const int* __restrict__ src, const int* __restrict__ dst,
                                 float* degO, float* degI, int nE) {
    int e = blockIdx.x * blockDim.x + threadIdx.x;
    if (e < nE) {
        unsafeAtomicAdd(&degO[src[e]], 1.0f);
        unsafeAtomicAdd(&degI[dst[e]], 1.0f);
    }
}

__global__ void rsqrt_kernel(float* p, int n) {
    int i = blockIdx.x * blockDim.x + threadIdx.x;
    if (i < n) p[i] = rsqrtf(p[i]);
}

// Pre-swizzle a [K x 128] fp32 matrix into per-wave B fragments (bf16 hi/lo split).
// Layout: [kb][nt][lane][r] dwords.  B 32x16 bf16 fragment: n=lane%16, h=lane/16;
// r<4 -> K=8h+2r,+1; r>=4 -> K=16+8h+2(r-4),+1 (two bf16 packed per dword, low=even K).
__global__ void pack_b_kernel(const float* __restrict__ B, int K, int KB,
                              unsigned* __restrict__ hi, unsigned* __restrict__ lo) {
    int idx = blockIdx.x * blockDim.x + threadIdx.x;
    int total = KB * 2048;   // 8 nt * 32 lanes * 8 dwords
    if (idx >= total) return;
    int r    = idx & 7;
    int lane = (idx >> 3) & 31;
    int nt   = (idx >> 8) & 7;
    int kb   = idx >> 11;
    int n    = nt * 16 + (lane & 15);
    int kk   = ((r & 4) ? 16 : 0) + ((lane >> 4) * 8) + 2 * (r & 3);
    int k0   = kb * 32 + kk;
    float x0 = (k0     < K) ? B[(size_t)k0 * 128 + n]       : 0.0f;
    float x1 = (k0 + 1 < K) ? B[(size_t)(k0 + 1) * 128 + n] : 0.0f;
    unsigned dhi, dlo;
    split2(x0, x1, dhi, dlo);
    hi[idx] = dhi;
    lo[idx] = dlo;
}

// ---------------- WMMA GEMM:  out[M x 128] = A[M x K] @ B[K x 128] ----------------
// Block = 256 threads (8 waves), tile M=32 x N=128.  Wave w owns N-tile w (16 cols)
// for BOTH 16-row M-subtiles, so each B fragment pair feeds 6 WMMAs.
// Split-precision: C += Ah*Bh + Ah*Bl + Al*Bh  (3x v_wmma_f32_16x16x32_bf16 per tile).
template <bool GUARD>
__device__ __forceinline__ void gemm_kb_step(
    const float* __restrict__ A, const unsigned* __restrict__ Bhi,
    const unsigned* __restrict__ Blo,
    unsigned* sAhi, unsigned* sAlo,
    int kb, int K, int m0, int lane, int wave,
    int row, int c4, int sidx, FragC& acc0, FragC& acc1)
{
    int k = kb * 32 + c4;
    float4 a4 = make_float4(0.f, 0.f, 0.f, 0.f);
    if (!GUARD || (k < K)) a4 = *(const float4*)(A + (size_t)(m0 + row) * K + k);
    unsigned d0h, d0l, d1h, d1l;
    split2(a4.x, a4.y, d0h, d0l);
    split2(a4.z, a4.w, d1h, d1l);
    sAhi[sidx] = d0h; sAhi[sidx + 1] = d1h;
    sAlo[sidx] = d0l; sAlo[sidx + 1] = d1l;
    __syncthreads();

    Frag16 ah0, al0, ah1, al1;
    ah0.q[0] = *(const uint4*)&sAhi[lane * 8];
    ah0.q[1] = *(const uint4*)&sAhi[lane * 8 + 4];
    al0.q[0] = *(const uint4*)&sAlo[lane * 8];
    al0.q[1] = *(const uint4*)&sAlo[lane * 8 + 4];
    ah1.q[0] = *(const uint4*)&sAhi[256 + lane * 8];
    ah1.q[1] = *(const uint4*)&sAhi[256 + lane * 8 + 4];
    al1.q[0] = *(const uint4*)&sAlo[256 + lane * 8];
    al1.q[1] = *(const uint4*)&sAlo[256 + lane * 8 + 4];

    size_t boff = ((size_t)(kb * 8 + wave) * 32 + lane) * 8;
    Frag16 bh, bl;
    bh.q[0] = *(const uint4*)(Bhi + boff);
    bh.q[1] = *(const uint4*)(Bhi + boff + 4);
    bl.q[0] = *(const uint4*)(Blo + boff);
    bl.q[1] = *(const uint4*)(Blo + boff + 4);

    acc0.v = __builtin_amdgcn_wmma_f32_16x16x32_bf16(false, ah0.v, false, bh.v, (short)0, acc0.v, false, false);
    acc0.v = __builtin_amdgcn_wmma_f32_16x16x32_bf16(false, ah0.v, false, bl.v, (short)0, acc0.v, false, false);
    acc0.v = __builtin_amdgcn_wmma_f32_16x16x32_bf16(false, al0.v, false, bh.v, (short)0, acc0.v, false, false);
    acc1.v = __builtin_amdgcn_wmma_f32_16x16x32_bf16(false, ah1.v, false, bh.v, (short)0, acc1.v, false, false);
    acc1.v = __builtin_amdgcn_wmma_f32_16x16x32_bf16(false, ah1.v, false, bl.v, (short)0, acc1.v, false, false);
    acc1.v = __builtin_amdgcn_wmma_f32_16x16x32_bf16(false, al1.v, false, bh.v, (short)0, acc1.v, false, false);
    __syncthreads();
}

__global__ __launch_bounds__(256)
void wmma_gemm_kernel(const float* __restrict__ A,
                      const unsigned* __restrict__ Bhi, const unsigned* __restrict__ Blo,
                      const float* __restrict__ rowScale,
                      float* __restrict__ out0, float* __restrict__ out1,
                      int K, int KB) {
    __shared__ __align__(16) unsigned sAhi[512];   // two 16x32 bf16 tiles (hi)
    __shared__ __align__(16) unsigned sAlo[512];   // two 16x32 bf16 tiles (lo)
    const int tid  = threadIdx.x;
    const int lane = tid & 31;
    const int wave = tid >> 5;          // 0..7 -> N-tile
    const int m0   = blockIdx.x * 32;

    // A staging: thread loads float4 at (row, c4..c4+3) of the 32x32 tile.
    const int row = tid >> 3;           // 0..31
    const int c4  = (tid & 7) * 4;
    // (row%16, klocal) -> A-fragment (lane, vgpr) per ISA 16-bit A 16x32 layout.
    const int lane_off = (c4 & 8) ? 16 : 0;
    const int rbase    = ((c4 & 16) ? 4 : 0) + ((c4 & 7) >> 1);
    const int sidx     = ((row >> 4) << 8) + ((row & 15) + lane_off) * 8 + rbase;

    FragC acc0, acc1;
#pragma unroll
    for (int j = 0; j < 8; ++j) { acc0.f[j] = 0.0f; acc1.f[j] = 0.0f; }

    const int KBfull = K >> 5;          // unguarded 32-wide K blocks
    for (int kb = 0; kb < KBfull; ++kb)
        gemm_kb_step<false>(A, Bhi, Blo, sAhi, sAlo, kb, K, m0, lane, wave,
                            row, c4, sidx, acc0, acc1);
    if (KBfull < KB)                    // K tail (only last block partially valid)
        gemm_kb_step<true>(A, Bhi, Blo, sAhi, sAlo, KBfull, K, m0, lane, wave,
                           row, c4, sidx, acc0, acc1);

    // C/D layout: VGPR j -> (M=j, N=lane) lanes 0-15; (M=j+8, N=lane-16) lanes 16-31.
    const int n = wave * 16 + (lane & 15);
#pragma unroll
    for (int t = 0; t < 2; ++t) {
        const FragC& acc = t ? acc1 : acc0;
#pragma unroll
        for (int j = 0; j < 8; ++j) {
            int m = m0 + t * 16 + j + ((lane >> 4) << 3);
            float v = acc.f[j];
            if (rowScale) v *= rowScale[m];
            out0[(size_t)m * 128 + n] = v;
            if (out1) out1[(size_t)m * 128 + n] = v;
        }
    }
}

// ---------------- SpMM edge aggregation (agg pre-seeded with self-loop h) ----------------
__global__ __launch_bounds__(256)
void spmm_kernel(const int* __restrict__ src, const int* __restrict__ dst,
                 const float* __restrict__ h1, const float* __restrict__ h2,
                 float* __restrict__ a1, float* __restrict__ a2, int nE) {
    int e = blockIdx.x * 2 + (threadIdx.x >> 7);
    int c = threadIdx.x & 127;
    if (e < nE) {
        int s = src[e], d = dst[e];
        unsafeAtomicAdd(&a1[(size_t)d * 128 + c], h1[(size_t)s * 128 + c]);
        unsafeAtomicAdd(&a2[(size_t)d * 128 + c], h2[(size_t)s * 128 + c]);
    }
}

// ---------------- uv = [relu(c)+relu(t) | c+t]  (element-wise epilogue of both GraphConvs) ----
__global__ __launch_bounds__(256)
void build_uv_kernel(const float* __restrict__ agg1, const float* __restrict__ agg2,
                     const float* __restrict__ rsIn,
                     const float* __restrict__ b_conv, const float* __restrict__ conv_w,
                     const float* __restrict__ b_fus,  const float* __restrict__ td_w,
                     float* __restrict__ uv, int total) {
    int idx = blockIdx.x * 256 + threadIdx.x;    // over nRows*128
    if (idx >= total) return;
    int i = idx >> 7, j = idx & 127;
    float ri = rsIn[i];
    float c = (agg1[idx] * ri + b_conv[j]) * conv_w[j];
    float t = (agg2[idx] * ri + b_fus[j]) * td_w[j];
    uv[(size_t)i * 256 + j]       = fmaxf(c, 0.0f) + fmaxf(t, 0.0f);
    uv[(size_t)i * 256 + 128 + j] = c + t;
}

// ---------------- bias + LayerNorm over the WMMA result ----------------
__global__ __launch_bounds__(128)
void ln_kernel(const float* __restrict__ res, const float* __restrict__ catB,
               const float* __restrict__ gamma, const float* __restrict__ beta,
               float* __restrict__ out) {
    __shared__ float red[128];
    const int i = blockIdx.x, j = threadIdx.x;
    float acc = res[(size_t)i * 128 + j] + catB[j];
    red[j] = acc; __syncthreads();
    for (int s = 64; s > 0; s >>= 1) { if (j < s) red[j] += red[j + s]; __syncthreads(); }
    float mean = red[0] * (1.0f / 128.0f);
    __syncthreads();
    float d = acc - mean;
    red[j] = d * d; __syncthreads();
    for (int s = 64; s > 0; s >>= 1) { if (j < s) red[j] += red[j + s]; __syncthreads(); }
    float var = red[0] * (1.0f / 128.0f);
    out[(size_t)i * 128 + j] = d * rsqrtf(var + 1e-5f) * gamma[j] + beta[j];
}

extern "C" void kernel_launch(void* const* d_in, const int* in_sizes, int n_in,
                              void* d_out, int out_size, void* d_ws, size_t ws_size,
                              hipStream_t stream) {
    (void)n_in; (void)out_size; (void)ws_size;
    const float* curr_h   = (const float*)d_in[0];
    const float* next_h   = (const float*)d_in[1];
    const float* curr_inc = (const float*)d_in[2];
    const int*   src      = (const int*)d_in[3];
    const int*   dst      = (const int*)d_in[4];
    const float* W_conv   = (const float*)d_in[5];
    const float* b_conv   = (const float*)d_in[6];
    const float* W_fus    = (const float*)d_in[7];
    const float* b_fus    = (const float*)d_in[8];
    const float* conv_w   = (const float*)d_in[9];
    const float* td_w     = (const float*)d_in[10];
    const float* cat_W    = (const float*)d_in[11];
    const float* cat_b    = (const float*)d_in[12];
    const float* gamma    = (const float*)d_in[13];
    const float* beta     = (const float*)d_in[14];

    const int D      = 128;
    const int nCurr  = in_sizes[0] / D;   // 20000
    const int nNext  = in_sizes[1] / D;   // 10000
    const int nE     = in_sizes[3];       // 320000
    const int KBbig  = (nNext + 31) / 32; // 313
    const int KBfeat = D / 32;            // 4
    const int KBcat  = (2 * D) / 32;      // 8

    float* ws = (float*)d_ws;
    size_t off = 0;
    float* rsO   = ws + off; off += nCurr;               // deg_out -> rsqrt
    float* rsI   = ws + off; off += nCurr;               // deg_in  -> rsqrt
    float* fused = ws + off; off += (size_t)nCurr * D;
    float* h1    = ws + off; off += (size_t)nCurr * D;
    float* h2    = ws + off; off += (size_t)nCurr * D;
    float* agg1  = ws + off; off += (size_t)nCurr * D;
    float* agg2  = ws + off; off += (size_t)nCurr * D;
    float* uv    = ws + off; off += (size_t)nCurr * 2 * D;
    float* res   = ws + off; off += (size_t)nCurr * D;
    unsigned* Bhi  = (unsigned*)(ws + off); off += (size_t)KBbig * 2048;
    unsigned* Blo  = (unsigned*)(ws + off); off += (size_t)KBbig * 2048;
    unsigned* WhiC = (unsigned*)(ws + off); off += (size_t)KBfeat * 2048;
    unsigned* WloC = (unsigned*)(ws + off); off += (size_t)KBfeat * 2048;
    unsigned* WhiF = (unsigned*)(ws + off); off += (size_t)KBfeat * 2048;
    unsigned* WloF = (unsigned*)(ws + off); off += (size_t)KBfeat * 2048;
    unsigned* CWhi = (unsigned*)(ws + off); off += (size_t)KBcat * 2048;
    unsigned* CWlo = (unsigned*)(ws + off); off += (size_t)KBcat * 2048;

    // Degrees (init to 1.0 for the self loop), then rsqrt in place.
    fill_kernel<<<(2 * nCurr + 255) / 256, 256, 0, stream>>>(rsO, 1.0f, 2 * nCurr);
    deg_count_kernel<<<(nE + 255) / 256, 256, 0, stream>>>(src, dst, rsO, rsI, nE);
    rsqrt_kernel<<<(2 * nCurr + 255) / 256, 256, 0, stream>>>(rsO, 2 * nCurr);

    // Pre-swizzle all B operands into WMMA fragment layout (bf16 hi/lo split).
    pack_b_kernel<<<(KBbig * 2048 + 255) / 256, 256, 0, stream>>>(next_h, nNext, KBbig, Bhi, Blo);
    pack_b_kernel<<<(KBfeat * 2048 + 255) / 256, 256, 0, stream>>>(W_conv, D, KBfeat, WhiC, WloC);
    pack_b_kernel<<<(KBfeat * 2048 + 255) / 256, 256, 0, stream>>>(W_fus,  D, KBfeat, WhiF, WloF);
    pack_b_kernel<<<(KBcat * 2048 + 255) / 256, 256, 0, stream>>>(cat_W, 2 * D, KBcat, CWhi, CWlo);

    // Big GEMM: fused = curr_inc @ next_h   (M=20000, K=10000, N=128)
    wmma_gemm_kernel<<<nCurr / 32, 256, 0, stream>>>(curr_inc, Bhi, Blo, nullptr,
                                                     fused, nullptr, nNext, KBbig);
    // h1 = (curr_h @ W_conv) * rsqrt(deg_out); agg1 seeded = h1 (self loop)
    wmma_gemm_kernel<<<nCurr / 32, 256, 0, stream>>>(curr_h, WhiC, WloC, rsO,
                                                     h1, agg1, D, KBfeat);
    // h2 = (fused @ W_fus) * rsqrt(deg_out); agg2 seeded = h2
    wmma_gemm_kernel<<<nCurr / 32, 256, 0, stream>>>(fused, WhiF, WloF, rsO,
                                                     h2, agg2, D, KBfeat);

    // Edge aggregation
    spmm_kernel<<<(nE + 1) / 2, 256, 0, stream>>>(src, dst, h1, h2, agg1, agg2, nE);

    // Epilogue: uv = [relu|skip], res = uv @ cat_W (WMMA), then bias + LayerNorm
    build_uv_kernel<<<(nCurr * D + 255) / 256, 256, 0, stream>>>(agg1, agg2, rsI, b_conv, conv_w,
                                                                 b_fus, td_w, uv, nCurr * D);
    wmma_gemm_kernel<<<nCurr / 32, 256, 0, stream>>>(uv, CWhi, CWlo, nullptr,
                                                     res, nullptr, 2 * D, KBcat);
    ln_kernel<<<nCurr, 128, 0, stream>>>(res, cat_b, gamma, beta, (float*)d_out);
}
